// MambaModel_86543591014777
// MI455X (gfx1250) — matrix operand order repaired and checked by reading, not compiled, not run on hardware
//
#include <hip/hip_runtime.h>
#include <hip/hip_bf16.h>

// ---------------------------------------------------------------------------
// Mamba forward for MI455X (gfx1250), wave32, WMMA f16->f32 matrix cores.
// ---------------------------------------------------------------------------

typedef __attribute__((ext_vector_type(16))) _Float16 v16h;
typedef __attribute__((ext_vector_type(8)))  float    v8f;

#define THREADS 256

// ------------------------- RMSNorm (per row) -------------------------------
__global__ __launch_bounds__(THREADS)
void k_rmsnorm(const float* __restrict__ x, const float* __restrict__ w,
               float* __restrict__ out, int D)
{
    __shared__ float red[THREADS];
    const int row = blockIdx.x;
    const int tid = threadIdx.x;
    const float* xr = x + (size_t)row * D;
    float s = 0.f;
    for (int i = tid; i < D; i += THREADS) { float v = xr[i]; s += v * v; }
    red[tid] = s; __syncthreads();
    for (int o = THREADS / 2; o > 0; o >>= 1) {
        if (tid < o) red[tid] += red[tid + o];
        __syncthreads();
    }
    const float rms = rsqrtf(red[0] / (float)D + 1e-6f);
    float* orow = out + (size_t)row * D;
    for (int i = tid; i < D; i += THREADS) orow[i] = xr[i] * rms * w[i];
}

// ------------------- WMMA GEMM: C[M,N] = A[M,K] * W[N,K]^T -----------------
// Each wave computes a (16*MSUB)M x (16*NSUB)N tile with MSUB*NSUB f32
// accumulators; all tile loops are compile-time so control flow around the
// WMMAs is branch-free (EXEC stays all-1s, no accumulator shuffling).
// f32 inputs are converted inline to f16 (v_cvt_pk_f16_f32), accumulation is
// f32 via v_wmma_f32_16x16x32_f16. RES fuses a residual add in the epilogue.
// Requires: M % (16*MSUB) == 0, N % (16*NSUB) == 0, K % 32 == 0.
template <int MSUB, int NSUB, bool RES>
__global__ __launch_bounds__(THREADS)
void k_gemm_wmma(const float* __restrict__ A, const float* __restrict__ W,
                 const float* __restrict__ res, float* __restrict__ C,
                 int M, int N, int K)
{
    const int lane = threadIdx.x;            // 0..31 (wave32)
    const int wave = threadIdx.y;            // 0..7
    const int tiles_n = N / (16 * NSUB);
    const int tiles_m = M / (16 * MSUB);
    const int tile = blockIdx.x * 8 + wave;
    if (tile >= tiles_m * tiles_n) return;   // wave-uniform exit
    const int tm = tile / tiles_n;
    const int tn = tile % tiles_n;
    const int m0 = tm * 16 * MSUB;
    const int n0 = tn * 16 * NSUB;

    const int row = lane & 15;
    const int hi  = lane >> 4;               // 0 or 1

    v8f acc[MSUB][NSUB];
    #pragma unroll
    for (int s = 0; s < MSUB; ++s)
        #pragma unroll
        for (int j = 0; j < NSUB; ++j) acc[s][j] = (v8f){};

    for (int kk = 0; kk < K; kk += 32) {
        // ---- A fragments (16x32 f16): ISA layout
        //   element i -> K = kk + i + (i>=8 ? 8 : 0) + (lane>=16 ? 8 : 0)
        v16h a[MSUB];
        #pragma unroll
        for (int s = 0; s < MSUB; ++s) {
            const float* ap = A + (size_t)(m0 + s * 16 + row) * K + kk + hi * 8;
            #pragma unroll
            for (int i = 0; i < 8; ++i) {
                a[s][i]     = (_Float16)ap[i];
                a[s][i + 8] = (_Float16)ap[i + 16];
            }
            if (kk + 32 < K) __builtin_prefetch(ap + 32, 0, 3);   // global_prefetch
        }
        // ---- B fragments (32x16 f16), B[k][n] = W[n*K + k]
        //   element i -> K = kk + i + (i>=8 ? 8 : 0) + hi*8 ; column N = n0 + j*16 + (lane&15)
        #pragma unroll
        for (int j = 0; j < NSUB; ++j) {
            const float* wp = W + (size_t)(n0 + j * 16 + row) * K + kk + hi * 8;
            v16h bm;
            #pragma unroll
            for (int i = 0; i < 16; ++i) bm[i] = (_Float16)wp[i + (i >= 8 ? 8 : 0)];
            if (kk + 32 < K) __builtin_prefetch(wp + 32, 0, 3);
            #pragma unroll
            for (int s = 0; s < MSUB; ++s)
                acc[s][j] = __builtin_amdgcn_wmma_f32_16x16x32_f16(
                    /*neg_a=*/false, a[s], /*neg_b=*/false, bm,
                    /*c_mod=*/(short)0, acc[s][j],
                    /*reuse_a=*/false, /*reuse_b=*/false);
        }
    }

    // ---- D layout: VGPR r -> M = r + (lane>=16 ? 8 : 0) ; N = lane&15
    #pragma unroll
    for (int s = 0; s < MSUB; ++s)
        #pragma unroll
        for (int j = 0; j < NSUB; ++j)
            #pragma unroll
            for (int r = 0; r < 8; ++r) {
                const int mm = m0 + s * 16 + r + hi * 8;
                const int nn = n0 + j * 16 + row;
                const size_t idx = (size_t)mm * N + nn;
                float v = acc[s][j][r];
                if (RES) v += res[idx];
                C[idx] = v;
            }
}

// ----------------- depthwise causal conv (k=4) + bias + SiLU ---------------
__global__ __launch_bounds__(THREADS)
void k_conv_silu(const float* __restrict__ xz, const float* __restrict__ cw,
                 const float* __restrict__ cb, float* __restrict__ xc,
                 int L, int DI, int K2, int total)
{
    const int idx = blockIdx.x * THREADS + threadIdx.x;
    if (idx >= total) return;
    const int d = idx % DI;
    const int l = (idx / DI) % L;
    const int b = idx / (DI * L);
    float acc = cb[d];
    #pragma unroll
    for (int k = 0; k < 4; ++k) {
        const int lp = l - 3 + k;
        if (lp >= 0)
            acc += cw[d * 4 + k] * xz[(size_t)(b * L + lp) * K2 + d];
    }
    const float sg = 1.f / (1.f + expf(-acc));
    xc[(size_t)idx] = acc * sg;
}

// ----------------- delta = softplus(dt @ dt_w^T + dt_b) --------------------
__global__ __launch_bounds__(THREADS)
void k_dt_proj(const float* __restrict__ xdbl, const float* __restrict__ dtw,
               const float* __restrict__ dtb, float* __restrict__ delta,
               int DI, int P, int R)
{
    __shared__ float dtr[64];
    const int m = blockIdx.y;
    const int d = blockIdx.x * THREADS + threadIdx.x;
    if (threadIdx.x < R) dtr[threadIdx.x] = xdbl[(size_t)m * P + threadIdx.x];
    __syncthreads();
    float acc = dtb[d];
    #pragma unroll 8
    for (int r = 0; r < R; ++r) acc += dtr[r] * dtw[(size_t)d * R + r];
    const float sp = (acc > 20.f) ? acc : log1pf(expf(acc));
    delta[(size_t)m * DI + d] = sp;
}

// ----------------- selective scan + D-skip + SiLU(z) gating ----------------
// One thread per (b, d) channel; 16 hidden states in registers.
__global__ __launch_bounds__(THREADS)
void k_scan(const float* __restrict__ delta, const float* __restrict__ xc,
            const float* __restrict__ xdbl, const float* __restrict__ xz,
            const float* __restrict__ A_log, const float* __restrict__ Dsk,
            float* __restrict__ y, int L, int DI, int P, int K2)
{
    const int d = blockIdx.x * THREADS + threadIdx.x;
    const int b = blockIdx.y;
    float a[16], h[16];
    #pragma unroll
    for (int n = 0; n < 16; ++n) {
        a[n] = -expf(A_log[(size_t)d * 16 + n]);
        h[n] = 0.f;
    }
    const float dskip = Dsk[d];
    for (int t = 0; t < L; ++t) {
        const size_t rowp = (size_t)(b * L + t);
        const float dl = delta[rowp * DI + d];
        const float u  = xc[rowp * DI + d];
        const float* Bs = xdbl + rowp * P + 48;
        const float* Cs = xdbl + rowp * P + 64;
        const float du = dl * u;
        float yv = 0.f;
        #pragma unroll
        for (int n = 0; n < 16; ++n) {
            const float dA = expf(dl * a[n]);
            h[n] = dA * h[n] + du * Bs[n];
            yv += h[n] * Cs[n];
        }
        yv += u * dskip;
        const float zv = xz[rowp * K2 + DI + d];
        const float sg = 1.f / (1.f + expf(-zv));
        y[rowp * DI + d] = yv * (zv * sg);
    }
}

// --------- head: rmsnorm(last) -> h1 -> rmsnorm -> gelu -> h2 --------------
__global__ __launch_bounds__(THREADS)
void k_head(const float* __restrict__ xin, const float* __restrict__ fnw,
            const float* __restrict__ h1w, const float* __restrict__ h1b,
            const float* __restrict__ hnw, const float* __restrict__ h2w,
            const float* __restrict__ h2b, float* __restrict__ out,
            int L, int D, int NC)
{
    __shared__ float red[THREADS];
    __shared__ float ln[768];
    __shared__ float hb[768];
    const int b = blockIdx.x, tid = threadIdx.x;
    const float* xl = xin + ((size_t)b * L + (L - 1)) * D;

    float s = 0.f;
    for (int i = tid; i < D; i += THREADS) { float v = xl[i]; s += v * v; }
    red[tid] = s; __syncthreads();
    for (int o = THREADS / 2; o > 0; o >>= 1) { if (tid < o) red[tid] += red[tid + o]; __syncthreads(); }
    const float rms = rsqrtf(red[0] / (float)D + 1e-6f);
    for (int i = tid; i < D; i += THREADS) ln[i] = xl[i] * rms * fnw[i];
    __syncthreads();

    for (int j = tid; j < D; j += THREADS) {
        float acc = h1b[j];
        const float* wr = h1w + (size_t)j * D;
        for (int k = 0; k < D; ++k) acc += ln[k] * wr[k];
        hb[j] = acc;
    }
    __syncthreads();

    s = 0.f;
    for (int i = tid; i < D; i += THREADS) { float v = hb[i]; s += v * v; }
    red[tid] = s; __syncthreads();
    for (int o = THREADS / 2; o > 0; o >>= 1) { if (tid < o) red[tid] += red[tid + o]; __syncthreads(); }
    const float rms2 = rsqrtf(red[0] / (float)D + 1e-6f);
    for (int i = tid; i < D; i += THREADS) {
        const float v = hb[i] * rms2 * hnw[i];
        ln[i] = 0.5f * v * (1.f + erff(v * 0.70710678118f));  // exact GELU
    }
    __syncthreads();

    for (int j = tid; j < NC; j += THREADS) {
        float acc = h2b[j];
        const float* wr = h2w + (size_t)j * D;
        for (int k = 0; k < D; ++k) acc += ln[k] * wr[k];
        out[(size_t)b * NC + j] = acc;
    }
}

// ---------------------------------------------------------------------------
extern "C" void kernel_launch(void* const* d_in, const int* in_sizes, int n_in,
                              void* d_out, int out_size, void* d_ws, size_t ws_size,
                              hipStream_t stream)
{
    const float* x      = (const float*)d_in[0];
    const float* norm_w = (const float*)d_in[1];
    const float* in_w   = (const float*)d_in[2];
    const float* conv_w = (const float*)d_in[3];
    const float* conv_b = (const float*)d_in[4];
    const float* xp_w   = (const float*)d_in[5];
    const float* dt_w   = (const float*)d_in[6];
    const float* dt_b   = (const float*)d_in[7];
    const float* A_log  = (const float*)d_in[8];
    const float* D_skip = (const float*)d_in[9];
    const float* out_w  = (const float*)d_in[10];
    const float* fn_w   = (const float*)d_in[11];
    const float* h1w    = (const float*)d_in[12];
    const float* h1b    = (const float*)d_in[13];
    const float* hnw    = (const float*)d_in[14];
    const float* h2w    = (const float*)d_in[15];
    const float* h2b    = (const float*)d_in[16];
    (void)in_sizes; (void)n_in; (void)out_size; (void)ws_size;

    constexpr int Bsz = 2, L = 1024, D = 768, DI = 1536, K2 = 3072, P = 80, R = 48, NC = 10;
    const int M = Bsz * L;  // 2048

    // workspace layout (≈83 MB, 256B aligned)
    char* base = (char*)d_ws;
    size_t off = 0;
    auto alloc = [&](size_t bytes) -> char* {
        char* p = base + off;
        off = (off + bytes + 255) & ~(size_t)255;
        return p;
    };
    float* xn    = (float*)alloc((size_t)M * D  * 4);
    float* xz    = (float*)alloc((size_t)M * K2 * 4);
    float* xc    = (float*)alloc((size_t)M * DI * 4);
    float* xdbl  = (float*)alloc((size_t)M * P  * 4);
    float* delta = (float*)alloc((size_t)M * DI * 4);
    float* ybuf  = (float*)alloc((size_t)M * DI * 4);
    float* bufA  = (float*)alloc((size_t)M * D  * 4);
    float* bufB  = (float*)alloc((size_t)M * D  * 4);

    const float* cur = x;   // layer input / residual source
    float* nxt = bufA;

    for (int i = 0; i < 4; ++i) {
        k_rmsnorm<<<M, THREADS, 0, stream>>>(cur, norm_w + i * D, xn, D);

        { // in_proj: (2048 x 3072) = xn(2048x768) @ in_w^T ; 32x64 wave tiles
            int tiles = (M / 32) * (K2 / 64), blocks = (tiles + 7) / 8;
            k_gemm_wmma<2, 4, false><<<blocks, dim3(32, 8), 0, stream>>>(
                xn, in_w + (size_t)i * K2 * D, nullptr, xz, M, K2, D);
        }
        {
            int total = M * DI;
            k_conv_silu<<<(total + THREADS - 1) / THREADS, THREADS, 0, stream>>>(
                xz, conv_w + i * DI * 4, conv_b + i * DI, xc, L, DI, K2, total);
        }
        { // x_proj: (2048 x 80) = xc(2048x1536) @ xp_w^T ; 32x80 wave tiles
            int tiles = (M / 32) * (P / 80), blocks = (tiles + 7) / 8;
            k_gemm_wmma<2, 5, false><<<blocks, dim3(32, 8), 0, stream>>>(
                xc, xp_w + (size_t)i * P * DI, nullptr, xdbl, M, P, DI);
        }
        k_dt_proj<<<dim3(DI / THREADS, M), THREADS, 0, stream>>>(
            xdbl, dt_w + (size_t)i * DI * R, dt_b + i * DI, delta, DI, P, R);

        k_scan<<<dim3(DI / THREADS, Bsz), THREADS, 0, stream>>>(
            delta, xc, xdbl, xz, A_log + (size_t)i * DI * 16, D_skip + i * DI,
            ybuf, L, DI, P, K2);

        { // out_proj + residual: (2048 x 768) = y(2048x1536) @ out_w^T + cur
            int tiles = (M / 32) * (D / 64), blocks = (tiles + 7) / 8;
            k_gemm_wmma<2, 4, true><<<blocks, dim3(32, 8), 0, stream>>>(
                ybuf, out_w + (size_t)i * D * DI, cur, nxt, M, D, DI);
        }
        cur = nxt;
        nxt = (nxt == bufA) ? bufB : bufA;
    }

    k_head<<<Bsz, THREADS, 0, stream>>>(cur, fn_w, h1w, h1b, hnw, h2w, h2b,
                                        (float*)d_out, L, D, NC);
}
